// WindowedContextCA_38491496906758
// MI455X (gfx1250) — matrix operand order repaired and checked by reading, not compiled, not run on hardware
//
#include <hip/hip_runtime.h>

// ---------------------------------------------------------------------------
// WindowedContextCA on MI455X (gfx1250, wave32, WMMA).
//
// Roofline: ~88 GFLOP vs ~0.4 GB HBM traffic -> at 23.3 TB/s the memory floor
// is ~17us, so all matrix stages go through v_wmma_f32_16x16x32_f16 (16384
// FLOP/instr, 8x the f32 WMMA rate), f16 operands + fp32 accumulation.
// Attention V-staging uses GLOBAL_LOAD_ASYNC_TO_LDS_B128 (ASYNCcnt) when the
// toolchain exposes the builtins, with a plain-load fallback otherwise.
// ---------------------------------------------------------------------------

#define CDIM  256
#define DDIM  32
#define HEADS 8
#define HW    128
#define TOK   256     // 16*16 tokens per window
#define NWIN  256     // 4 * 8 * 8 windows

typedef __attribute__((ext_vector_type(16))) _Float16 v16h;
typedef __attribute__((ext_vector_type(8)))  _Float16 v8h;
typedef __attribute__((ext_vector_type(8)))  float    v8f;
typedef __attribute__((ext_vector_type(4)))  float    v4f;

#if __has_builtin(__builtin_amdgcn_global_load_async_to_lds_b128) && \
    __has_builtin(__builtin_amdgcn_s_wait_asynccnt)
#define HAVE_ASYNC_LDS 1
// Probe-derived signature: arg0 = v4i32 in AS(1) (global src, non-const),
// arg1 = v4i32 in AS(3) (LDS dst), arg2 = imm offset, arg3 = imm cpol.
typedef __attribute__((ext_vector_type(4))) int v4i_;
typedef __attribute__((address_space(1))) v4i_* gas_ptr;
typedef __attribute__((address_space(3))) v4i_* las_ptr;
#endif

// 16 contiguous halfs (32B-aligned) -> full B fragment for one 16-lane half.
__device__ __forceinline__ v16h ld16(const _Float16* p){ return *(const v16h*)p; }

// Two 8-half chunks (16B-aligned each) -> A fragment in contiguous k order:
// lane<16 holds K {0..15}, lane>=16 holds K {16..31}, same order as ld16 B.
__device__ __forceinline__ v16h ld8x2(const _Float16* p0, const _Float16* p1){
  v8h a = *(const v8h*)p0;
  v8h b = *(const v8h*)p1;
  v16h r;
#pragma unroll
  for (int e = 0; e < 8; ++e){ r[e] = a[e]; r[e+8] = b[e]; }
  return r;
}

// Same as ld8x2 but from fp32 source with on-the-fly f32->f16 conversion.
__device__ __forceinline__ v16h ldf32x16(const float* p0, const float* p1){
  v4f a0 = *(const v4f*)p0, a1 = *(const v4f*)(p0+4);
  v4f b0 = *(const v4f*)p1, b1 = *(const v4f*)(p1+4);
  v16h r;
#pragma unroll
  for (int e = 0; e < 4; ++e){
    r[e]    = (_Float16)a0[e];
    r[e+4]  = (_Float16)a1[e];
    r[e+8]  = (_Float16)b0[e];
    r[e+12] = (_Float16)b1[e];
  }
  return r;
}

__device__ __forceinline__ v8f wmma16(v16h a, v16h b, v8f c){
  // D = A(16x32 f16) * B(32x16 f16) + C(16x16 f32)
  return __builtin_amdgcn_wmma_f32_16x16x32_f16(false, a, false, b,
                                                (short)0, c, false, false);
}

// ---------------------------------------------------------------------------
// Kernel 0: weight conversion fp32->f16 with transpose to N-major/K-contig.
// ---------------------------------------------------------------------------
__global__ __launch_bounds__(256) void prep_weights(
    const float* __restrict__ qw, const float* __restrict__ outw,
    const float* __restrict__ kvw,
    _Float16* __restrict__ qw_t, _Float16* __restrict__ outw_t,
    _Float16* __restrict__ kvw_t){
  const int i = blockIdx.x * 256 + threadIdx.x;
  if (i < CDIM * CDIM){
    const int k = i >> 8, n = i & 255;              // src (K,N) row-major
    qw_t  [n * CDIM + k] = (_Float16)qw[i];         // dst (N,K)
    outw_t[n * CDIM + k] = (_Float16)outw[i];
  }
  if (i < 25 * 32 * 512){                           // kv_w (5,5,32,512)
    const int tap = i / (32 * 512);
    const int rem = i - tap * (32 * 512);
    const int k = rem >> 9;
    const int o = rem & 511;
    kvw_t[(tap * 512 + o) * 32 + k] = (_Float16)kvw[i]; // (tap, o, k)
  }
}

// ---------------------------------------------------------------------------
// Kernel 1: Q = x_wsa @ q_w + q_b, scaled by 1/sqrt(d), written to the
// windowed [win][head][token][d] f16 layout consumed by attention.
// One wave = 16 rows x 64 cols tile; K loop 8 x 32.
// ---------------------------------------------------------------------------
__global__ __launch_bounds__(256) void qproj_kernel(
    const float* __restrict__ x, const float* __restrict__ qb,
    const _Float16* __restrict__ qw_t, _Float16* __restrict__ q_ws){
  const int lane = threadIdx.x & 31, wid = threadIdx.x >> 5;
  const int hi = lane >> 4, ln = lane & 15;
  const int task   = blockIdx.x * 8 + wid;     // 16384 tasks
  const int mtile  = task >> 2;
  const int nchunk = task & 3;
  v8f acc[4] = {};
  const float* arow = x + (size_t)(mtile * 16 + ln) * CDIM;
#pragma unroll
  for (int kb = 0; kb < 8; ++kb){
    if (kb < 7) __builtin_prefetch(arow + (kb + 1) * 32, 0, 3);
    const float* p0 = arow + kb * 32 + 16 * hi;
    const v16h a = ldf32x16(p0, p0 + 8);
#pragma unroll
    for (int nt = 0; nt < 4; ++nt){
      const int n = nchunk * 64 + nt * 16 + ln;
      acc[nt] = wmma16(a, ld16(qw_t + n * CDIM + kb * 32 + 16 * hi), acc[nt]);
    }
  }
  const float scale = 0.17677669529663687f;  // 1/sqrt(32), folded into q
#pragma unroll
  for (int r = 0; r < 8; ++r){
    const int row = mtile * 16 + r + 8 * hi;   // D-layout: M = r + 8*(lane>>4)
    const int bb = row >> 14, yy = (row >> 7) & 127, xx = row & 127;
    const int win   = bb * 64 + (yy >> 4) * 8 + (xx >> 4);
    const int token = (yy & 15) * 16 + (xx & 15);
#pragma unroll
    for (int nt = 0; nt < 4; ++nt){
      const int c = nchunk * 64 + nt * 16 + ln;
      const float v = (acc[nt][r] + qb[c]) * scale;
      q_ws[(((size_t)(win * HEADS + (c >> 5)) * TOK + token) << 5) + (c & 31)] =
          (_Float16)v;
    }
  }
}

// ---------------------------------------------------------------------------
// Kernel 2: grouped 5x5 conv (pad 2) as 25 shifted K=32 GEMMs with WMMA.
// One wave = 16 x-positions (one window row) x one group's 64 output chans.
// Outputs routed into windowed k/v f16 buffers (+kv_b).
// ---------------------------------------------------------------------------
__global__ __launch_bounds__(256) void kvconv_kernel(
    const float* __restrict__ x, const float* __restrict__ kvb,
    const _Float16* __restrict__ kvw_t,
    _Float16* __restrict__ k_ws, _Float16* __restrict__ v_ws){
  const int lane = threadIdx.x & 31, wid = threadIdx.x >> 5;
  const int hi = lane >> 4, ln = lane & 15;
  const int task = blockIdx.x * 8 + wid;       // 32768 tasks
  const int g  = task & 7;                     // conv group
  const int mt = task >> 3;                    // (b, y, x0/16)
  const int x0 = (mt & 7) * 16;
  const int y  = (mt >> 3) & 127;
  const int bb = mt >> 10;
  v8f acc[4] = {};
  const int xl = x0 + ln;                      // this lane's M position (x)
#pragma unroll
  for (int dy = 0; dy < 5; ++dy){
    const int yy = y + dy - 2;
    if (yy < 0 || yy >= HW) continue;          // uniform: whole tap is zero
    const float* srow = x + (size_t)(bb * HW + yy) * HW * CDIM + g * 32;
#pragma unroll
    for (int dx = 0; dx < 5; ++dx){
      const int xs = xl + dx - 2;
      v16h a = {};
      if (xs >= 0 && xs < HW){                 // per-lane padding; reconverges
        const float* p0 = srow + (size_t)xs * CDIM + 16 * hi;
        a = ldf32x16(p0, p0 + 8);
      }
      const int tap = dy * 5 + dx;
      const _Float16* wt =
          kvw_t + (size_t)(tap * 512 + g * 64 + ln) * 32 + 16 * hi;
#pragma unroll
      for (int nt = 0; nt < 4; ++nt){
        acc[nt] = wmma16(a, ld16(wt + (size_t)nt * 16 * 32), acc[nt]);
      }
    }
  }
  const int win = bb * 64 + (y >> 4) * 8 + (x0 >> 4);
#pragma unroll
  for (int r = 0; r < 8; ++r){
    const int M = r + 8 * hi;
    const int token = (y & 15) * 16 + M;
#pragma unroll
    for (int nt = 0; nt < 4; ++nt){
      const int o = g * 64 + nt * 16 + ln;
      const float v = acc[nt][r] + kvb[o];
      if (o < 256){
        k_ws[(((size_t)(win * HEADS + (o >> 5)) * TOK + token) << 5) + (o & 31)] =
            (_Float16)v;
      } else {
        const int c = o - 256;
        v_ws[(((size_t)(win * HEADS + (c >> 5)) * TOK + token) << 5) + (c & 31)] =
            (_Float16)v;
      }
    }
  }
}

// ---------------------------------------------------------------------------
// Kernel 3: windowed attention. One block (4 waves) per (window, head).
// Scores: 16 WMMAs per 16-query tile (K=32=d). Rel-pos bias gathered from
// the 30KB table. Softmax: lane-local + 16-lane shfl_xor reductions (the two
// wave32 halves hold disjoint row sets per the D layout). P is re-fragmented
// through LDS; V is staged via async-to-LDS (ASYNCcnt) then transposed for
// contiguous B fragments. Static LDS: 48 KB (stage buffer overlays attn_s).
// ---------------------------------------------------------------------------
__global__ __launch_bounds__(128) void attn_kernel(
    const _Float16* __restrict__ q_ws, const _Float16* __restrict__ k_ws,
    const _Float16* __restrict__ v_ws, const float* __restrict__ bias_table,
    _Float16* __restrict__ o_ws){
  __shared__ __align__(32) _Float16 vt[DDIM * TOK];        // 16 KB, V^T
  __shared__ __align__(32) _Float16 attn_s[4][16 * TOK];   // 32 KB, per-wave P
  const int lane = threadIdx.x & 31, wid = threadIdx.x >> 5;
  const int hi = lane >> 4, ln = lane & 15;
  const int win = blockIdx.x >> 3, head = blockIdx.x & 7;
  const size_t base = (size_t)(win * HEADS + head) * TOK * DDIM;
  const _Float16* qp = q_ws + base;
  const _Float16* kp = k_ws + base;
  const _Float16* vp = v_ws + base;

#if defined(HAVE_ASYNC_LDS)
  // Stage the 16KB V tile row-major into LDS with async b128 copies
  // (8 issues x 128 lanes x 16B), overlaying attn_s (unused until later).
  _Float16* stage = &attn_s[0][0];
#pragma unroll
  for (int it = 0; it < 8; ++it){
    const int idx = it * 128 + (int)threadIdx.x;           // 16B chunk id
    __builtin_amdgcn_global_load_async_to_lds_b128(
        (gas_ptr)(vp + idx * 8), (las_ptr)(stage + idx * 8), 0, 0);
  }
  __builtin_amdgcn_s_wait_asynccnt(0);   // this wave's copies done
  __syncthreads();                       // => everyone's copies done
  for (int t = threadIdx.x; t < TOK; t += 128){            // LDS transpose
    const _Float16* vr = stage + t * DDIM;
#pragma unroll
    for (int d = 0; d < DDIM; ++d) vt[d * TOK + t] = vr[d];
  }
#else
  for (int t = threadIdx.x; t < TOK; t += 128){            // stage V^T
    const _Float16* vrow = vp + t * DDIM;
#pragma unroll
    for (int d = 0; d < DDIM; ++d) vt[d * TOK + t] = vrow[d];
  }
#endif
  __syncthreads();

  const int wbat = win >> 6, wy = (win >> 3) & 7, wx = win & 7;
  for (int mi = 0; mi < 4; ++mi){
    const int mtile = wid * 4 + mi;                        // 16 query rows
    const _Float16* qrow = qp + (mtile * 16 + ln) * DDIM + 16 * hi;
    const v16h aq = ld8x2(qrow, qrow + 8);
    v8f s[16];
#pragma unroll
    for (int nt = 0; nt < 16; ++nt){                       // S = Q K^T (scaled)
      v8f z = {};
      s[nt] = wmma16(aq, ld16(kp + (nt * 16 + ln) * DDIM + 16 * hi), z);
    }
    // bias + softmax; ktok = nt*16 + ln  =>  ky = nt, kx = ln
#pragma unroll
    for (int r = 0; r < 8; ++r){
      const int qtok = mtile * 16 + r + 8 * hi;
      const int qy = qtok >> 4, qx = qtok & 15;
      float mx = -3.0e38f;
#pragma unroll
      for (int nt = 0; nt < 16; ++nt){
        const int rel = (qy - nt + 15) * 31 + (qx - ln + 15);
        const float val = s[nt][r] + bias_table[rel * HEADS + head];
        s[nt][r] = val;
        mx = fmaxf(mx, val);
      }
#pragma unroll
      for (int m = 1; m < 16; m <<= 1) mx = fmaxf(mx, __shfl_xor(mx, m, 32));
      float sum = 0.0f;
#pragma unroll
      for (int nt = 0; nt < 16; ++nt){
        const float e = __expf(s[nt][r] - mx);
        s[nt][r] = e;
        sum += e;
      }
#pragma unroll
      for (int m = 1; m < 16; m <<= 1) sum += __shfl_xor(sum, m, 32);
      const float inv = 1.0f / sum;
      _Float16* prow = &attn_s[wid][(r + 8 * hi) * TOK + ln];
#pragma unroll
      for (int nt = 0; nt < 16; ++nt) prow[nt * 16] = (_Float16)(s[nt][r] * inv);
    }
    // O = P @ V  (same-wave LDS RAW; hardware keeps DS ops in order)
    v8f o0 = {}, o1 = {};
#pragma unroll
    for (int kb = 0; kb < 8; ++kb){
      const _Float16* ap = &attn_s[wid][ln * TOK + kb * 32 + 16 * hi];
      const v16h pa = ld8x2(ap, ap + 8);
      o0 = wmma16(pa, ld16(&vt[(ln     ) * TOK + kb * 32 + 16 * hi]), o0);
      o1 = wmma16(pa, ld16(&vt[(16 + ln) * TOK + kb * 32 + 16 * hi]), o1);
    }
#pragma unroll
    for (int r = 0; r < 8; ++r){                           // un-window write
      const int tok = mtile * 16 + r + 8 * hi;
      const int yy = wy * 16 + (tok >> 4), xxp = wx * 16 + (tok & 15);
      _Float16* dst =
          o_ws + ((size_t)(wbat * HW + yy) * HW + xxp) * CDIM + head * DDIM;
      dst[ln]      = (_Float16)o0[r];
      dst[16 + ln] = (_Float16)o1[r];
    }
  }
}

// ---------------------------------------------------------------------------
// Kernel 4: out = rezero * (O @ out_w + out_b), fp32 output.
// ---------------------------------------------------------------------------
__global__ __launch_bounds__(256) void outproj_kernel(
    const _Float16* __restrict__ o_ws, const float* __restrict__ outb,
    const _Float16* __restrict__ outw_t, const float* __restrict__ rezero,
    float* __restrict__ out){
  const int lane = threadIdx.x & 31, wid = threadIdx.x >> 5;
  const int hi = lane >> 4, ln = lane & 15;
  const int task   = blockIdx.x * 8 + wid;
  const int mtile  = task >> 2;
  const int nchunk = task & 3;
  v8f acc[4] = {};
  const _Float16* arow = o_ws + (size_t)(mtile * 16 + ln) * CDIM;
#pragma unroll
  for (int kb = 0; kb < 8; ++kb){
    if (kb < 7) __builtin_prefetch(arow + (kb + 1) * 32, 0, 3);
    const _Float16* p0 = arow + kb * 32 + 16 * hi;
    const v16h a = ld8x2(p0, p0 + 8);
#pragma unroll
    for (int nt = 0; nt < 4; ++nt){
      const int n = nchunk * 64 + nt * 16 + ln;
      acc[nt] = wmma16(a, ld16(outw_t + n * CDIM + kb * 32 + 16 * hi), acc[nt]);
    }
  }
  const float rz = rezero[0];
#pragma unroll
  for (int r = 0; r < 8; ++r){
    const int row = mtile * 16 + r + 8 * hi;
#pragma unroll
    for (int nt = 0; nt < 4; ++nt){
      const int c = nchunk * 64 + nt * 16 + ln;
      out[(size_t)row * CDIM + c] = rz * (acc[nt][r] + outb[c]);
    }
  }
}

// ---------------------------------------------------------------------------
// Workspace layout (bytes):
//   qw_t    @ 0          131072   (256x256 f16, N-major)
//   outw_t  @ 131072     131072
//   kvw_t   @ 262144     819200   (25 taps x 512 out-ch x 32 in-ch f16)
//   q_ws    @ 1081344    33554432 ([win][head][token][d] f16)
//   k_ws    @ 34635776   33554432
//   v_ws    @ 68190208   33554432
//   o_ws    @ 101744640  33554432 ((B,H,W,C) f16)
//   total   = 135299072 bytes (~135.3 MB)
// ---------------------------------------------------------------------------
extern "C" void kernel_launch(void* const* d_in, const int* in_sizes, int n_in,
                              void* d_out, int out_size, void* d_ws, size_t ws_size,
                              hipStream_t stream){
  (void)in_sizes; (void)n_in; (void)out_size; (void)ws_size;
  const float* x_wsa  = (const float*)d_in[0];
  const float* x_orig = (const float*)d_in[1];
  const float* q_w    = (const float*)d_in[2];
  const float* q_b    = (const float*)d_in[3];
  const float* kv_w   = (const float*)d_in[4];
  const float* kv_b   = (const float*)d_in[5];
  const float* out_w  = (const float*)d_in[6];
  const float* out_b  = (const float*)d_in[7];
  const float* bias_t = (const float*)d_in[8];
  const float* rezero = (const float*)d_in[9];
  float* out = (float*)d_out;

  char* w = (char*)d_ws;
  _Float16* qw_t   = (_Float16*)(w + 0);
  _Float16* outw_t = (_Float16*)(w + 131072);
  _Float16* kvw_t  = (_Float16*)(w + 262144);
  _Float16* q_ws   = (_Float16*)(w + 1081344);
  _Float16* k_ws   = (_Float16*)(w + 34635776);
  _Float16* v_ws   = (_Float16*)(w + 68190208);
  _Float16* o_ws   = (_Float16*)(w + 101744640);

  prep_weights  <<<1600, 256, 0, stream>>>(q_w, out_w, kv_w, qw_t, outw_t, kvw_t);
  qproj_kernel  <<<2048, 256, 0, stream>>>(x_wsa, q_b, qw_t, q_ws);
  kvconv_kernel <<<4096, 256, 0, stream>>>(x_orig, kv_b, kvw_t, k_ws, v_ws);
  attn_kernel   <<<2048, 128, 0, stream>>>(q_ws, k_ws, v_ws, bias_t, o_ws);
  outproj_kernel<<<2048, 256, 0, stream>>>(o_ws, out_b, outw_t, rezero, out);
}
